// PCNN_31834297598410
// MI455X (gfx1250) — matrix operand order repaired and checked by reading, not compiled, not run on hardware
//
#include <hip/hip_runtime.h>

typedef __attribute__((ext_vector_type(16))) __bf16 v16bf;
typedef __attribute__((ext_vector_type(8)))  float  v8f;
typedef __attribute__((ext_vector_type(8)))  __bf16 v8bf;
typedef __attribute__((ext_vector_type(4)))  __bf16 v4bf;

#define BB 64
#define LL 512
#define MM (BB*LL)      // 32768
#define DE 768
#define DP 320
#define KC 2304         // 3*DE, conv reduction dim
#define KTB 24576       // 768*32: elems per 32-wide k-tile of a swizzled weight

// workspace byte offsets (all 256-aligned)
#define OFF_W1B  0u           // 768*768 bf16   = 1179648 B
#define OFF_W2B  1179648u     // 320*768 bf16   =  491520 B
#define OFF_WCB  1703936u     // 2304*768 bf16  = 3538944 B
#define OFF_XB   5242880u     // 32768*768 bf16 = 50331648 B
#define OFF_POOL 55574528u    // 64*768*3 i32   =  589824 B

// ---------------------------------------------------------------------------
// CDNA5 async global->LDS copy (ASYNCcnt) via inline asm (VGLOBAL encoding:
// VDST = wave-relative LDS byte address, VADDR = 64-bit global address).
// Generic pointers to __shared__ truncate to the LDS offset (ISA 10.2).
// Set USE_ASYNC 0 to fall back to sync copies.
// ---------------------------------------------------------------------------
#define USE_ASYNC 1

__device__ inline void cp_async_b128(const __bf16* gsrc, __bf16* ldst) {
#if USE_ASYNC
  const unsigned int laddr = (unsigned int)(unsigned long long)(size_t)ldst;
  const unsigned long long gaddr = (unsigned long long)(size_t)gsrc;
  asm volatile("global_load_async_to_lds_b128 %0, %1, off"
               :: "v"(laddr), "v"(gaddr)
               : "memory");
#else
  *(v8bf*)ldst = *(const v8bf*)gsrc;
#endif
}

__device__ inline void wait_async() {
#if USE_ASYNC
  asm volatile("s_wait_asynccnt 0x0" ::: "memory");
#endif
}

// ---------------------------------------------------------------------------
// Fragment loaders (contiguous-k order, same in A and B).  Both A and B tiles are stored in
// LDS with the reduction dim contiguous, [row][40] bf16 (80B stride: every
// 8-elem async chunk is 16B-aligned; 20-bank stride avoids conflicts).
// A frag: lane halves hold K 0..15 / 16..31.
// B frag: lane halves hold K 0..15 / 16..31 -> one contiguous 32B read.
// ---------------------------------------------------------------------------
__device__ inline v16bf load_frag_a40(const __bf16* Abase, int mrow, int lane) {
  const int r  = lane & 15;
  const int kb = (lane >> 4) << 4;
  const __bf16* ap = Abase + (size_t)(mrow + r) * 40;
  v16bf f;
#pragma unroll
  for (int v = 0; v < 8; ++v) {
    const int k = kb + (v << 1);
    f[2*v]   = ap[k];
    f[2*v+1] = ap[k+1];
  }
  return f;
}

__device__ inline v16bf load_frag_b40(const __bf16* Bbase, int lane, int nf) {
  const int n  = (lane & 15) + (nf << 4);
  const int kb = (lane >> 4) << 4;
  const __bf16* bp = Bbase + (size_t)n * 40 + kb;
  v16bf f;
#pragma unroll
  for (int v = 0; v < 16; ++v) f[v] = bp[v];
  return f;
}

// ---------------------------------------------------------------------------
// Weight prep: swizzle to fragment-ready tiled bf16 layout
//   dst[(k/32)*KTB + n*32 + (k%32)] = B[k][n]
// so a 16B async chunk along k lands directly in the N-major LDS tile.
// ---------------------------------------------------------------------------
__global__ void prep_w1_kernel(const float* __restrict__ W1, __bf16* __restrict__ dst) {
  int i = blockIdx.x * 256 + threadIdx.x;
  if (i < 768 * 768) {
    int kt = i / KTB, rem = i - kt * KTB;
    int n = rem >> 5, kl = rem & 31;
    int k = (kt << 5) + kl;                    // B[k][n] = W1[n][k]
    dst[i] = (__bf16)W1[(size_t)n * 768 + k];
  }
}
__global__ void prep_w2_kernel(const float* __restrict__ W2, __bf16* __restrict__ dst) {
  int i = blockIdx.x * 256 + threadIdx.x;
  if (i < 320 * 768) {
    int kt = i / KTB, rem = i - kt * KTB;
    int n = rem >> 5, kl = rem & 31;
    int k = (kt << 5) + kl;                    // B[k][n] = W2[n][k], K=320
    dst[i] = (__bf16)W2[(size_t)n * 320 + k];
  }
}
__global__ void prep_wc_kernel(const float* __restrict__ Wc, __bf16* __restrict__ dst) {
  int i = blockIdx.x * 256 + threadIdx.x;
  if (i < KC * 768) {
    int kt = i / KTB, rem = i - kt * KTB;
    int n = rem >> 5, kl = rem & 31;
    int k = (kt << 5) + kl;                    // k = t*768 + ic
    int t = k / 768, ic = k - t * 768;
    dst[i] = (__bf16)Wc[(size_t)n * KC + ic * 3 + t];  // Wc[n][ic][t]
  }
}
__global__ void zero_pool_kernel(int* __restrict__ pooled) {
  int i = blockIdx.x * 256 + threadIdx.x;
  if (i < BB * 768 * 3) pooled[i] = 0;
}

// ---------------------------------------------------------------------------
// Kernel 1: fused gate.  Block tile 128(M) x 64(N), 256 threads = 8 waves,
// wave w -> rows [w*16, w*16+16).  acc1 = Xe*W1^T (K=768), acc2 = Xp*W2^T
// (K=320), X = sigmoid(acc1+b1)*Xe + (1-sig)*tanh(acc2+b2) -> Xb (bf16).
// A staged fp32->bf16 by VALU; B tiles async-DMA'd to LDS.
// ---------------------------------------------------------------------------
__global__ __launch_bounds__(256) void gate_kernel(
    const float* __restrict__ Xe, const float* __restrict__ Xp,
    const __bf16* __restrict__ W1t, const __bf16* __restrict__ W2t,
    const float* __restrict__ b1, const float* __restrict__ b2,
    __bf16* __restrict__ Xb)
{
  __shared__ __bf16 As[128][40];
  __shared__ __bf16 Bs[64][40];
  const int tid  = threadIdx.x;
  const int lane = tid & 31, w = tid >> 5;
  const int m0 = blockIdx.x << 7;
  const int n0 = blockIdx.y << 6;
  const int rowA = tid >> 1;            // 0..127
  const int cbA  = (tid & 1) << 4;      // 0 / 16
  const int nB   = tid >> 2;            // 0..63
  const int cB   = (tid & 3) << 3;      // 0,8,16,24 (one 8-elem chunk)
  const int mw   = w << 4;

  v8f acc1[4] = {};
  v8f acc2[4] = {};

  // ---- GEMM 1: A = Xe (fp32 -> bf16), B = W1t, K = 768
#pragma unroll 1
  for (int k0 = 0; k0 < DE; k0 += 32) {
    __syncthreads();
    {
      const float* src = Xe + (size_t)(m0 + rowA) * DE + k0 + cbA;
      __bf16* d = &As[rowA][cbA];
#pragma unroll
      for (int q = 0; q < 16; q += 4) {
        float4 vv = *(const float4*)(src + q);
        v4bf o; o[0]=(__bf16)vv.x; o[1]=(__bf16)vv.y; o[2]=(__bf16)vv.z; o[3]=(__bf16)vv.w;
        *(v4bf*)(d + q) = o;
      }
      if (k0 + 32 < DE) __builtin_prefetch(src + 32, 0, 3);
    }
    cp_async_b128(W1t + (size_t)(k0 >> 5) * KTB + (size_t)(n0 + nB) * 32 + cB,
                  &Bs[nB][cB]);
    wait_async();
    __syncthreads();
    const v16bf a = load_frag_a40(&As[0][0], mw, lane);
#pragma unroll
    for (int f = 0; f < 4; ++f) {
      const v16bf b = load_frag_b40(&Bs[0][0], lane, f);
      acc1[f] = __builtin_amdgcn_wmma_f32_16x16x32_bf16(
          false, a, false, b, (short)0, acc1[f], false, false);
    }
  }

  // ---- GEMM 2: A = Xp (fp32 -> bf16), B = W2t, K = 320
#pragma unroll 1
  for (int k0 = 0; k0 < DP; k0 += 32) {
    __syncthreads();
    {
      const float* src = Xp + (size_t)(m0 + rowA) * DP + k0 + cbA;
      __bf16* d = &As[rowA][cbA];
#pragma unroll
      for (int q = 0; q < 16; q += 4) {
        float4 vv = *(const float4*)(src + q);
        v4bf o; o[0]=(__bf16)vv.x; o[1]=(__bf16)vv.y; o[2]=(__bf16)vv.z; o[3]=(__bf16)vv.w;
        *(v4bf*)(d + q) = o;
      }
      if (k0 + 32 < DP) __builtin_prefetch(src + 32, 0, 3);
    }
    cp_async_b128(W2t + (size_t)(k0 >> 5) * KTB + (size_t)(n0 + nB) * 32 + cB,
                  &Bs[nB][cB]);
    wait_async();
    __syncthreads();
    const v16bf a = load_frag_a40(&As[0][0], mw, lane);
#pragma unroll
    for (int f = 0; f < 4; ++f) {
      const v16bf b = load_frag_b40(&Bs[0][0], lane, f);
      acc2[f] = __builtin_amdgcn_wmma_f32_16x16x32_bf16(
          false, a, false, b, (short)0, acc2[f], false, false);
    }
  }

  // ---- fused epilogue (C/D layout: lanes 0-15 n=lane,m=r; lanes 16-31 m=r+8)
#pragma unroll
  for (int r = 0; r < 8; ++r) {
    const int ml = mw + ((lane >> 4) << 3) + r;
    const int mg = m0 + ml;
#pragma unroll
    for (int f = 0; f < 4; ++f) {
      const int ng = n0 + (lane & 15) + (f << 4);
      const float g1 = acc1[f][r] + b1[ng];
      const float g2 = acc2[f][r] + b2[ng];
      const float av = 1.0f / (1.0f + __expf(-g1));
      const float xe = Xe[(size_t)mg * DE + ng];
      const float xv = av * xe + (1.0f - av) * tanhf(g2);
      Xb[(size_t)mg * DE + ng] = (__bf16)xv;
    }
  }
}

// ---------------------------------------------------------------------------
// Kernel 2: conv1d(k=3,same) as GEMM over K=2304, block tile 128x128,
// 256 threads = 8 waves, wave = 32(M) x 64(N) -> 8 WMMA / k-step with
// operand reuse.  Double-buffered LDS fed by async global->LDS B128 DMA.
// Epilogue: segment-masked max pool via positive-only int atomicMax.
// ---------------------------------------------------------------------------
__global__ __launch_bounds__(256) void conv_pool_kernel(
    const __bf16* __restrict__ Xb, const __bf16* __restrict__ Wct,
    const float* __restrict__ bc, const int* __restrict__ Xmask,
    int* __restrict__ pooled)
{
  __shared__ __bf16 As[2][128][40];
  __shared__ __bf16 Bs[2][128][40];
  const int tid  = threadIdx.x;
  const int lane = tid & 31, w = tid >> 5;
  const int m0 = blockIdx.x << 7;
  const int n0 = blockIdx.y << 7;
  const int bBase = m0 & ~(LL - 1);
  const int rowA = tid >> 1;            // 0..127
  const int cbA  = (tid & 1) << 4;      // 0 / 16  (two 8-elem chunks each)
  const int nB   = tid >> 1;            // 0..127
  const int cB   = (tid & 1) << 4;      // 0 / 16  (two 8-elem chunks each)
  const int mw   = (w >> 1) << 5;       // 0,32,64,96
  const int nw   = (w & 1) << 6;        // 0,64

  v8f acc[2][4] = {};

  auto issue = [&](int kk, int p) {
    const int t   = kk / 768;           // tap 0..2
    const int kin = kk - t * 768;
    const int srow = m0 + rowA + t - 1; // shifted row, zero outside batch
    __bf16* dA = &As[p][rowA][cbA];
    if (srow >= bBase && srow < bBase + LL) {
      const __bf16* sA = Xb + (size_t)srow * DE + kin + cbA;
      cp_async_b128(sA,     dA);
      cp_async_b128(sA + 8, dA + 8);
    } else {
      v8bf z = {};
      *(v8bf*)dA = z; *(v8bf*)(dA + 8) = z;
    }
    const __bf16* sB = Wct + (size_t)(kk >> 5) * KTB + (size_t)(n0 + nB) * 32 + cB;
    __bf16* dB = &Bs[p][nB][cB];
    cp_async_b128(sB,     dB);
    cp_async_b128(sB + 8, dB + 8);
  };

  issue(0, 0);
  wait_async();
  __syncthreads();

#pragma unroll 1
  for (int kk = 0; kk < KC; kk += 32) {
    const int p = (kk >> 5) & 1;
    if (kk + 32 < KC) issue(kk + 32, p ^ 1);

    const v16bf a0 = load_frag_a40(&As[p][0][0], mw,      lane);
    const v16bf a1 = load_frag_a40(&As[p][0][0], mw + 16, lane);
#pragma unroll
    for (int f = 0; f < 4; ++f) {
      const v16bf b = load_frag_b40(&Bs[p][0][0], lane, (nw >> 4) + f);
      acc[0][f] = __builtin_amdgcn_wmma_f32_16x16x32_bf16(
          false, a0, false, b, (short)0, acc[0][f], false, false);
      acc[1][f] = __builtin_amdgcn_wmma_f32_16x16x32_bf16(
          false, a1, false, b, (short)0, acc[1][f], false, false);
    }
    wait_async();
    __syncthreads();
  }

#pragma unroll
  for (int i = 0; i < 2; ++i) {
#pragma unroll
    for (int r = 0; r < 8; ++r) {
      const int ml = mw + (i << 4) + ((lane >> 4) << 3) + r;
      const int mg = m0 + ml;
      const int seg  = Xmask[mg];       // 0 = padding, 1..3 = segment
      const int bidx = mg >> 9;         // batch
#pragma unroll
      for (int f = 0; f < 4; ++f) {
        const int ng = n0 + nw + (lane & 15) + (f << 4);
        const float y = acc[i][f][r] + bc[ng];
        if (seg > 0 && y > 0.0f) {
          atomicMax(pooled + ((size_t)bidx * 768 + ng) * 3 + (seg - 1),
                    __float_as_int(y));
        }
      }
    }
  }
}

__global__ void finish_kernel(const int* __restrict__ pooled, float* __restrict__ out) {
  int i = blockIdx.x * 256 + threadIdx.x;
  if (i < BB * 768 * 3) out[i] = tanhf(__int_as_float(pooled[i]));
}

// ---------------------------------------------------------------------------
extern "C" void kernel_launch(void* const* d_in, const int* in_sizes, int n_in,
                              void* d_out, int out_size, void* d_ws, size_t ws_size,
                              hipStream_t stream) {
  (void)in_sizes; (void)n_in; (void)out_size; (void)ws_size;
  const float* Xp = (const float*)d_in[0];
  const float* Xe = (const float*)d_in[1];
  const int*   Xm = (const int*)  d_in[2];
  const float* W1 = (const float*)d_in[3];
  const float* b1 = (const float*)d_in[4];
  const float* W2 = (const float*)d_in[5];
  const float* b2 = (const float*)d_in[6];
  const float* Wc = (const float*)d_in[7];
  const float* bc = (const float*)d_in[8];
  float* out = (float*)d_out;

  char* ws = (char*)d_ws;
  __bf16* W1t = (__bf16*)(ws + OFF_W1B);
  __bf16* W2t = (__bf16*)(ws + OFF_W2B);
  __bf16* Wct = (__bf16*)(ws + OFF_WCB);
  __bf16* Xb  = (__bf16*)(ws + OFF_XB);
  int* pooled = (int*)   (ws + OFF_POOL);

  prep_w1_kernel<<<(768 * 768 + 255) / 256, 256, 0, stream>>>(W1, W1t);
  prep_w2_kernel<<<(320 * 768 + 255) / 256, 256, 0, stream>>>(W2, W2t);
  prep_wc_kernel<<<(KC * 768 + 255) / 256, 256, 0, stream>>>(Wc, Wct);
  zero_pool_kernel<<<(BB * 768 * 3 + 255) / 256, 256, 0, stream>>>(pooled);

  gate_kernel<<<dim3(MM / 128, DE / 64), 256, 0, stream>>>(Xe, Xp, W1t, W2t, b1, b2, Xb);
  conv_pool_kernel<<<dim3(MM / 128, DE / 128), 256, 0, stream>>>(Xb, Wct, bc, Xm, pooled);

  finish_kernel<<<(BB * 768 * 3 + 255) / 256, 256, 0, stream>>>(pooled, out);
}
